// SelfAttention_Module_multiscale_14499809591920
// MI455X (gfx1250) — matrix operand order repaired and checked by reading, not compiled, not run on hardware
//
#include <hip/hip_runtime.h>
#include <hip/hip_bf16.h>

// ---------------- geometry ----------------
#define NCLS   21
#define HW     41
#define WH     1681          // 41*41
#define PH     53            // padded image 53x53 (halo 6)
#define PRS    56            // padded row stride (elems), 16B-friendly
#define PP     2976          // padded channel stride (covers 52*56+59, mult of 16)
#define CIN    2048
#define COUT   1024
#define CHW    1089          // 33*33

typedef __attribute__((ext_vector_type(16))) __bf16 v16bf;
typedef __attribute__((ext_vector_type(8)))  float  v8f;

#if __has_builtin(__builtin_amdgcn_tensor_load_to_lds) && \
    __has_builtin(__builtin_amdgcn_s_wait_tensorcnt)
#define USE_TDM 1
typedef __attribute__((ext_vector_type(4))) unsigned int u32x4;
typedef __attribute__((ext_vector_type(8))) int          i32x8;
typedef __attribute__((ext_vector_type(4))) int          i32x4;
#endif

__device__ __forceinline__ unsigned short f2bf(float f) {
    unsigned u = __float_as_uint(f);
    unsigned r = u + 0x7FFFu + ((u >> 16) & 1u);   // RNE
    return (unsigned short)(r >> 16);
}

// ---- K1: bilinear 33->41 upsample into halo-padded bf16 image (zeros elsewhere)
__global__ void k_upsample(const float* __restrict__ x, unsigned short* __restrict__ xu)
{
    size_t idx = (size_t)blockIdx.x * blockDim.x + threadIdx.x;
    if (idx >= (size_t)4 * CIN * PP) return;
    int    p  = (int)(idx % PP);
    size_t ch = idx / PP;                 // b*2048+cin
    int pr = p / PRS, pc = p - pr * PRS;
    float val = 0.f;
    if (pr >= 6 && pr <= 46 && pc >= 6 && pc <= 46) {
        int oh = pr - 6, ow = pc - 6;
        float sy = fmaxf((oh + 0.5f) * (33.f / 41.f) - 0.5f, 0.f);
        float sx = fmaxf((ow + 0.5f) * (33.f / 41.f) - 0.5f, 0.f);
        int y0 = (int)sy, x0 = (int)sx;
        float wy = sy - (float)y0, wx = sx - (float)x0;
        int y1 = min(y0 + 1, 32), x1 = min(x0 + 1, 32);
        const float* xp = x + ch * CHW;
        float v00 = xp[y0 * 33 + x0], v01 = xp[y0 * 33 + x1];
        float v10 = xp[y1 * 33 + x0], v11 = xp[y1 * 33 + x1];
        val = v00 * (1.f - wy) * (1.f - wx) + v01 * (1.f - wy) * wx
            + v10 * wy * (1.f - wx)        + v11 * wy * wx;
    }
    xu[idx] = f2bf(val);
}

// ---- K2: weights [1024][2048][3][3] f32 -> [tap][cout][cin] bf16
__global__ void k_repack_w(const float* __restrict__ w, unsigned short* __restrict__ wr)
{
    size_t idx = (size_t)blockIdx.x * blockDim.x + threadIdx.x;
    if (idx >= (size_t)9 * COUT * CIN) return;
    int ci = (int)(idx % CIN);
    size_t t = idx / CIN;
    int co  = (int)(t % COUT);
    int tap = (int)(t / COUT);
    wr[idx] = f2bf(w[((size_t)co * CIN + ci) * 9 + tap]);
}

// ---- K3: dilated conv as implicit GEMM with V_WMMA_F32_16X16X32_BF16
// block = 128 thr (4 waves); block owns output row r, 64 couts, 48 (3x16) cols.
// Weights streamed per k-step by the Tensor Data Mover into LDS (when available).
__global__ void __launch_bounds__(128)
k_conv_wmma(const unsigned short* __restrict__ xu,
            const unsigned short* __restrict__ wr,
            const float* __restrict__ conv_b,
            float* __restrict__ f2)
{
    const int r    = blockIdx.x;            // output row 0..40
    const int cb   = blockIdx.y;            // cout block (64 couts)
    const int b    = blockIdx.z;
    const int tid  = threadIdx.x;
    const int wave = tid >> 5, lane = tid & 31;
    const int coutBase = cb * 64 + wave * 16;

    __shared__ unsigned short Xs[48 * 32];  // [pos][k], 3 KB, transposed activations
#ifdef USE_TDM
    __shared__ unsigned short Ws[64 * 32];  // [cout][k], 4 KB, TDM-staged weights
#endif

    const v8f vzero = {0.f,0.f,0.f,0.f,0.f,0.f,0.f,0.f};
    v8f acc0 = vzero, acc1 = vzero, acc2 = vzero;

    const int col = lane & 15;              // A: M row | B: N col | C: N col
    const int ab  = (lane >> 4) * 16;       // A k-offset (16 consecutive k)
    const int bb  = (lane >> 4) * 16;       // B k-offset (16 consecutive k)

#ifdef USE_TDM
    // D# group 1: data_size=2B, tile 32(k) x 64(cout), tensor==tile, stride0=2048
    i32x8 g1;
    g1[0] = 0x10000;        // [17:16] data_size = 1 (2 bytes)
    g1[1] = 32 << 16;       // tensor_dim0[15:0]
    g1[2] = 64 << 16;       // tensor_dim1[15:0]
    g1[3] = 32 << 16;       // tile_dim0 = 32
    g1[4] = 64;             // tile_dim1 = 64 (tile_dim2 = 0)
    g1[5] = 2048;           // tensor_dim0_stride[31:0]
    g1[6] = 0;
    g1[7] = 0;
    i32x4 gz = {0, 0, 0, 0};
#if defined(__clang_major__) && __clang_major__ >= 23
    i32x8 gz8 = {0, 0, 0, 0, 0, 0, 0, 0};
#endif
    const unsigned lds_ws = (unsigned)(size_t)(&Ws[0]);
#else
    const size_t wrow = ((size_t)(coutBase + col)) * CIN;  // per-lane A row (global path)
#endif

    for (int kh = 0; kh < 3; ++kh)
    for (int kw = 0; kw < 3; ++kw) {
        const size_t xbase = (size_t)b * CIN * PP + (size_t)(r + 6 * kh) * PRS + 6 * kw;
        const size_t wtap  = (size_t)(kh * 3 + kw) * COUT * CIN;
        for (int kc = 0; kc < CIN; kc += 32) {
            __syncthreads();                // protect LDS from previous iteration's readers
#ifdef USE_TDM
            if (wave == 0) {                // wave-uniform branch: single TDM issue
                size_t gaddr = (size_t)(wr + wtap + (size_t)(cb * 64) * CIN + kc);
                u32x4 g0;
                g0.x = 1u;                                           // count=1, user mode
                g0.y = lds_ws;                                       // LDS byte address
                g0.z = (unsigned)(gaddr & 0xFFFFFFFFu);              // global_addr[31:0]
                g0.w = (unsigned)((gaddr >> 32) & 0x01FFFFFFu) | 0x80000000u; // [56:32]|type=2
#if defined(__clang_major__) && __clang_major__ >= 23
                __builtin_amdgcn_tensor_load_to_lds(g0, g1, gz, gz, gz8, 0);
#else
                __builtin_amdgcn_tensor_load_to_lds(g0, g1, gz, gz, 0);
#endif
            }
#endif
            if (tid < 96) {                 // stage 32k x 48pos activations, transposed
                int k = tid / 3, third = tid - k * 3;
                const unsigned* src =
                    (const unsigned*)(xu + xbase + (size_t)(kc + k) * PP + third * 16);
                unsigned short* dst = &Xs[(third * 16) * 32 + k];
                #pragma unroll
                for (int j = 0; j < 8; ++j) {
                    unsigned v = src[j];
                    dst[(2 * j)     * 32] = (unsigned short)(v & 0xFFFFu);
                    dst[(2 * j + 1) * 32] = (unsigned short)(v >> 16);
                }
            }
#ifdef USE_TDM
            if (wave == 0) __builtin_amdgcn_s_wait_tensorcnt(0);
#endif
            __syncthreads();

            union { v16bf v; uint4 q[2]; } A;
#ifdef USE_TDM
            {   // A fragment from TDM-staged LDS: row = cout-in-block, two b128 reads
                const uint4* ap = (const uint4*)&Ws[(wave * 16 + col) * 32 + ab];
                A.q[0] = ap[0];             // k = ab .. ab+7
                A.q[1] = ap[1];             // k = ab+8 .. ab+15
            }
#else
            {
                __builtin_prefetch((const void*)(wr + wtap + wrow + kc + 32), 0, 1);
                const uint4* ap = (const uint4*)(wr + wtap + wrow + kc + ab);
                A.q[0] = ap[0];
                A.q[1] = ap[1];
            }
#endif
            union { v16bf v; uint4 q[2]; } B;
            const uint4* bp;
            bp = (const uint4*)&Xs[(0 * 16 + col) * 32 + bb]; B.q[0] = bp[0]; B.q[1] = bp[1];
            acc0 = __builtin_amdgcn_wmma_f32_16x16x32_bf16(false, A.v, false, B.v,
                                                           (short)0, acc0, false, false);
            bp = (const uint4*)&Xs[(1 * 16 + col) * 32 + bb]; B.q[0] = bp[0]; B.q[1] = bp[1];
            acc1 = __builtin_amdgcn_wmma_f32_16x16x32_bf16(false, A.v, false, B.v,
                                                           (short)0, acc1, false, false);
            bp = (const uint4*)&Xs[(2 * 16 + col) * 32 + bb]; B.q[0] = bp[0]; B.q[1] = bp[1];
            acc2 = __builtin_amdgcn_wmma_f32_16x16x32_bf16(false, A.v, false, B.v,
                                                           (short)0, acc2, false, false);
        }
    }

    const int mrow = (lane >> 4) * 8;       // C/D layout: col=lane&15, M=mrow+i
    v8f accs[3] = {acc0, acc1, acc2};
    #pragma unroll
    for (int nb = 0; nb < 3; ++nb) {
        int ow = nb * 16 + col;
        if (ow < HW) {
            #pragma unroll
            for (int i = 0; i < 8; ++i) {
                int co = coutBase + mrow + i;
                f2[((size_t)b * COUT + co) * WH + r * HW + ow] = accs[nb][i] + conv_b[co];
            }
        }
    }
}

// ---- K4: g[b][c][n] = sum_d cls_w[c][d] * f2[b][d][n]
__global__ void k_cls_proj(const float* __restrict__ f2, const float* __restrict__ clsw,
                           float* __restrict__ g)
{
    int idx = blockIdx.x * blockDim.x + threadIdx.x;
    if (idx >= 4 * NCLS * WH) return;
    int n = idx % WH;
    int t = idx / WH;
    int c = t % NCLS, b = t / NCLS;
    const float* fp = f2 + (size_t)b * COUT * WH + n;
    const float* wp = clsw + (size_t)c * COUT;
    float acc = 0.f;
    for (int d = 0; d < COUT; ++d) acc += wp[d] * fp[(size_t)d * WH];
    g[((size_t)b * NCLS + c) * WH + n] = acc;
}

// ---- K5: masked diffusion logits, all 4 scales in one pass over diffW
// wei_cum[b][s][c][m] = (sum_n g[c][n]*(diffW[m][n]>thr_s)) / cnt_s(m) + cls_b[c]
__global__ void __launch_bounds__(128)
k_diffuse(const float* __restrict__ diffW, const float* __restrict__ g,
          const float* __restrict__ ms, const float* __restrict__ cls_b,
          float* __restrict__ wei_cum)
{
    int bm = blockIdx.x;
    int b = bm / WH, m = bm - b * WH;
    int tid = threadIdx.x;
    if (tid >= 4 * NCLS) return;
    int s = tid / NCLS, c = tid - s * NCLS;
    float thr = ms[s];
    const float* dro = diffW + ((size_t)b * WH + m) * WH;
    const float* gr  = g + ((size_t)b * NCLS + c) * WH;
    float acc = 0.f, cnt = 0.f;
    for (int n = 0; n < WH; ++n) {
        float msk = (dro[n] > thr) ? 1.f : 0.f;
        cnt += msk;
        acc = fmaf(msk, gr[n], acc);
    }
    float logit = acc / fmaxf(cnt, 1.f);
    wei_cum[(((size_t)b * 4 + s) * NCLS + c) * WH + m] = logit + cls_b[c];
}

// ---- K6: per-pixel softmax chain; writes wei_mask, wei_cum2, wei outputs
__global__ void k_softmax_chain(const float* __restrict__ wei_cum,
                                const float* __restrict__ sm_w,
                                float* __restrict__ out_wc2,
                                float* __restrict__ out_wei,
                                float* __restrict__ out_mask)
{
    int idx = blockIdx.x * blockDim.x + threadIdx.x;
    if (idx >= 4 * WH) return;
    int b = idx / WH, m = idx - b * WH;
    float t[NCLS];
    #pragma unroll
    for (int c = 0; c < NCLS; ++c) {
        float w[4], sc[4];
        float mx = -1e30f;
        float smc = sm_w[c];
        #pragma unroll
        for (int s = 0; s < 4; ++s) {
            w[s]  = wei_cum[(((size_t)b * 4 + s) * NCLS + c) * WH + m];
            sc[s] = smc * w[s];
            mx = fmaxf(mx, sc[s]);
        }
        float e[4], sum = 0.f;
        #pragma unroll
        for (int s = 0; s < 4; ++s) { e[s] = __expf(sc[s] - mx); sum += e[s]; }
        float tc = 0.f;
        #pragma unroll
        for (int s = 0; s < 4; ++s) {
            float msk = e[s] / sum;
            out_mask[(((size_t)b * 4 + s) * NCLS + c) * WH + m] = msk;
            tc = fmaf(w[s], msk, tc);
        }
        t[c] = tc;
    }
    float mx = -1e30f;
    #pragma unroll
    for (int c = 0; c < NCLS; ++c) mx = fmaxf(mx, t[c]);
    float sum = 0.f, e2[NCLS];
    #pragma unroll
    for (int c = 0; c < NCLS; ++c) { e2[c] = __expf(t[c] - mx); sum += e2[c]; }
    float w2c[NCLS], mx2 = -1e30f;
    #pragma unroll
    for (int c = 0; c < NCLS; ++c) {
        w2c[c] = e2[c] / sum;
        out_wc2[((size_t)b * NCLS + c) * WH + m] = w2c[c];
        mx2 = fmaxf(mx2, w2c[c]);
    }
    float sum2 = 0.f, e3[NCLS];
    #pragma unroll
    for (int c = 0; c < NCLS; ++c) { e3[c] = __expf(w2c[c] - mx2); sum2 += e3[c]; }
    #pragma unroll
    for (int c = 0; c < NCLS; ++c)
        out_wei[((size_t)b * NCLS + c) * WH + m] = e3[c] / sum2;
}

// ---- K7: pooling + classifier: out[b][c] = sum_m softmax_m(wei*e^pw)[m] * g[c][m]
__global__ void __launch_bounds__(256)
k_pool_pred(const float* __restrict__ wei, const float* __restrict__ g,
            const float* __restrict__ pool_w, const float* __restrict__ cls_b,
            float* __restrict__ prob)
{
    int c = blockIdx.x, b = blockIdx.y;
    const float* wrow = wei + ((size_t)b * NCLS + c) * WH;
    const float* grow = g   + ((size_t)b * NCLS + c) * WH;
    float ep = __expf(pool_w[c]);
    __shared__ float red[256];
    int tid = threadIdx.x;
    float mx = -1e30f;
    for (int n = tid; n < WH; n += 256) mx = fmaxf(mx, wrow[n] * ep);
    red[tid] = mx; __syncthreads();
    for (int o = 128; o > 0; o >>= 1) { if (tid < o) red[tid] = fmaxf(red[tid], red[tid + o]); __syncthreads(); }
    mx = red[0]; __syncthreads();
    float sum = 0.f;
    for (int n = tid; n < WH; n += 256) sum += __expf(wrow[n] * ep - mx);
    red[tid] = sum; __syncthreads();
    for (int o = 128; o > 0; o >>= 1) { if (tid < o) red[tid] += red[tid + o]; __syncthreads(); }
    sum = red[0]; __syncthreads();
    float acc = 0.f;
    for (int n = tid; n < WH; n += 256) acc = fmaf(__expf(wrow[n] * ep - mx) / sum, grow[n], acc);
    red[tid] = acc; __syncthreads();
    for (int o = 128; o > 0; o >>= 1) { if (tid < o) red[tid] += red[tid + o]; __syncthreads(); }
    if (tid == 0) {
        float out = red[0] + cls_b[c];
        prob[b * NCLS + c] = 1.f / (1.f + __expf(-out));
    }
}

extern "C" void kernel_launch(void* const* d_in, const int* in_sizes, int n_in,
                              void* d_out, int out_size, void* d_ws, size_t ws_size,
                              hipStream_t stream)
{
    (void)in_sizes; (void)n_in; (void)out_size; (void)ws_size;
    const float* x      = (const float*)d_in[0];
    const float* diffW  = (const float*)d_in[1];
    const float* conv_w = (const float*)d_in[2];
    const float* conv_b = (const float*)d_in[3];
    const float* cls_w  = (const float*)d_in[4];
    const float* cls_b  = (const float*)d_in[5];
    const float* sm_w   = (const float*)d_in[6];
    const float* pool_w = (const float*)d_in[7];
    const float* ms     = (const float*)d_in[8];

    // workspace layout (bytes)
    char* ws = (char*)d_ws;
    unsigned short* xu  = (unsigned short*)(ws);                         // 4*2048*2976*2 = 48,758,784
    unsigned short* wr  = (unsigned short*)(ws + 48758784);              // 9*1024*2048*2 = 37,748,736
    float*          f2  = (float*)(ws + 48758784 + 37748736);            // 4*1024*1681*4 = 27,541,504
    float*          g   = (float*)(ws + 48758784 + 37748736 + 27541504); //   564,816
    float*          wcm = (float*)(ws + 48758784 + 37748736 + 27541504 + 564816); // 2,259,264
    // total ~116.9 MB

    float* prob     = (float*)d_out;                 // [4][21]       (as [4,1,21])
    float* wei_cum2 = prob + 4 * NCLS;               // [4][21][1681]
    float* wei      = wei_cum2 + 4 * NCLS * WH;      // [4][21][1681]
    float* wei_mask = wei + 4 * NCLS * WH;           // [4][4][21][1681]

    {   // weight repack
        size_t n = (size_t)9 * COUT * CIN;
        k_repack_w<<<dim3((unsigned)((n + 255) / 256)), dim3(256), 0, stream>>>(conv_w, wr);
    }
    {   // upsample -> padded bf16
        size_t n = (size_t)4 * CIN * PP;
        k_upsample<<<dim3((unsigned)((n + 255) / 256)), dim3(256), 0, stream>>>(x, xu);
    }
    // dilated conv via WMMA bf16 (+ TDM weight streaming)
    k_conv_wmma<<<dim3(HW, COUT / 64, 4), dim3(128), 0, stream>>>(xu, wr, conv_b, f2);
    // g = cls_w @ f2
    k_cls_proj<<<dim3((4 * NCLS * WH + 255) / 256), dim3(256), 0, stream>>>(f2, cls_w, g);
    // diffusion logits (all 4 scales)
    k_diffuse<<<dim3(4 * WH), dim3(128), 0, stream>>>(diffW, g, ms, cls_b, wcm);
    // softmax chain -> wei_mask, wei_cum2, wei
    k_softmax_chain<<<dim3((4 * WH + 127) / 128), dim3(128), 0, stream>>>(
        wcm, sm_w, wei_cum2, wei, wei_mask);
    // pooling + prediction -> prob
    k_pool_pred<<<dim3(NCLS, 4), dim3(256), 0, stream>>>(wei, g, pool_w, cls_b, prob);
}
